// SyntheticMoELayer_89026082112092
// MI455X (gfx1250) — matrix operand order repaired and checked by reading, not compiled, not run on hardware
//
#include <hip/hip_runtime.h>

// ---------------- problem constants ----------------
constexpr int T_TOK  = 8192;     // 4 * 2048 tokens
constexpr int H      = 2048;     // hidden
constexpr int IN     = 5632;     // intermediate
constexpr int NE     = 8;        // experts
constexpr int TILE_M = 128;      // grouped rows per tile
constexpr int TILE_N = 128;      // gate/up N tile
constexpr int TILE_ND = 256;     // down N tile
constexpr int BK     = 32;       // K per WMMA stage (bf16 16x16x32)
constexpr int LDW    = 20;       // LDS row stride in u32 (80B: 16B aligned, conflict-spread)
constexpr int PAD_CAP   = 2 * T_TOK + NE * TILE_M;   // 17408 grouped rows (padded)
constexpr int MAX_TILES = PAD_CAP / TILE_M;          // 136

// ---------------- workspace layout ----------------
constexpr size_t al256(size_t x) { return (x + 255) & ~size_t(255); }
constexpr size_t OFF_CW   = 0;                                        // [T,2] f32
constexpr size_t OFF_EIDX = al256(OFF_CW   + size_t(T_TOK) * 2 * 4);  // [T,2] i32
constexpr size_t OFF_POS  = al256(OFF_EIDX + size_t(T_TOK) * 2 * 4);  // [T,2] i32
constexpr size_t OFF_CNT  = al256(OFF_POS  + size_t(T_TOK) * 2 * 4);  // [8]
constexpr size_t OFF_CUR  = al256(OFF_CNT  + NE * 4);                 // [8]
constexpr size_t OFF_NT   = al256(OFF_CUR  + NE * 4);                 // [1]
constexpr size_t OFF_TE   = al256(OFF_NT   + 4);                      // [MAX_TILES]
constexpr size_t OFF_TR   = al256(OFF_TE   + size_t(MAX_TILES) * 4);  // [MAX_TILES]
constexpr size_t OFF_TL   = al256(OFF_TR   + size_t(MAX_TILES) * 4);  // [PAD_CAP]
constexpr size_t OFF_XB   = al256(OFF_TL   + size_t(PAD_CAP) * 4);    // [T,H] bf16
constexpr size_t OFF_ACT  = al256(OFF_XB   + size_t(T_TOK) * H * 2);  // [PAD_CAP, IN] bf16
constexpr size_t OFF_HOUT = al256(OFF_ACT  + size_t(PAD_CAP) * IN * 2); // [PAD_CAP, H] f32

// ---------------- types & helpers ----------------
typedef __attribute__((ext_vector_type(16))) __bf16 v16bf;
typedef __attribute__((ext_vector_type(8)))  float  v8f;

union FragBF { unsigned int u[8]; uint4 q[2]; v16bf v; };

__device__ inline unsigned int f32_to_bf16_bits(float f) {
  unsigned int u = __float_as_uint(f);
  return (u + 0x7FFFu + ((u >> 16) & 1u)) >> 16;   // round-to-nearest-even
}
__device__ inline unsigned int pack_bf16x2(float lo, float hi) {
  return f32_to_bf16_bits(lo) | (f32_to_bf16_bits(hi) << 16);
}
__device__ inline v8f zero8() {
  v8f z = {0.f, 0.f, 0.f, 0.f, 0.f, 0.f, 0.f, 0.f};
  return z;
}
// silu(g)*u with fast v_rcp_f32 (avoids IEEE div expansion in the epilogue)
__device__ inline float silu_mul(float g, float u) {
  return g * __builtin_amdgcn_rcpf(1.f + __expf(-g)) * u;
}

// LDS byte offset of a shared-memory pointer (flat LDS addresses use addr[31:0]).
__device__ __forceinline__ unsigned lds_off32(const void* p) {
  return (unsigned)(size_t)p;
}
// Async 16B copy global -> LDS (ASYNCcnt-tracked, no VGPR round trip).
__device__ __forceinline__ void async_copy_b128(unsigned lds_dst, const void* gsrc) {
  asm volatile("global_load_async_to_lds_b128 %0, %1, off"
               :: "v"(lds_dst), "v"(gsrc) : "memory");
}
__device__ __forceinline__ void wait_async0() {
  asm volatile("s_wait_asynccnt 0x0" ::: "memory");
}

// ---------------- kernel 0: zero counts ----------------
__global__ void moe_init_kernel(int* counts) {
  int i = threadIdx.x;
  if (i < NE) counts[i] = 0;
}

// ---------------- kernel 1: router (one wave per token) ----------------
__global__ __launch_bounds__(256) void moe_router_kernel(
    const float* __restrict__ x, const float* __restrict__ rw,
    const float* __restrict__ rb, float* __restrict__ cw,
    int* __restrict__ eidx, int* __restrict__ counts) {
  int wave = threadIdx.x >> 5, lane = threadIdx.x & 31;
  int t = blockIdx.x * 8 + wave;
  if (t >= T_TOK) return;
  const float* xr = x + (size_t)t * H;
  float acc[NE];
#pragma unroll
  for (int e = 0; e < NE; ++e) acc[e] = 0.f;
  for (int i = 0; i < H / 32; ++i) {
    int h = lane + i * 32;
    float xv = xr[h];
#pragma unroll
    for (int e = 0; e < NE; ++e) acc[e] += xv * rw[e * H + h];
  }
#pragma unroll
  for (int e = 0; e < NE; ++e)
    for (int o = 16; o > 0; o >>= 1) acc[e] += __shfl_xor(acc[e], o, 32);
  if (lane == 0) {
    float l[NE];
#pragma unroll
    for (int e = 0; e < NE; ++e) l[e] = acc[e] + rb[e];
    int i0 = 0;
#pragma unroll
    for (int e = 1; e < NE; ++e) if (l[e] > l[i0]) i0 = e;
    int i1 = -1;
#pragma unroll
    for (int e = 0; e < NE; ++e)
      if (e != i0 && (i1 < 0 || l[e] > l[i1])) i1 = e;
    float d = __expf(l[i1] - l[i0]);     // top-2 softmax renormalized
    float w0 = 1.f / (1.f + d);
    cw[2 * t + 0] = w0;
    cw[2 * t + 1] = 1.f - w0;
    eidx[2 * t + 0] = i0;
    eidx[2 * t + 1] = i1;
    atomicAdd(&counts[i0], 1);
    atomicAdd(&counts[i1], 1);
  }
}

// ---------------- kernel 2: setup tiles ----------------
__global__ void moe_setup_kernel(const int* __restrict__ counts, int* __restrict__ cursors,
                                 int* __restrict__ ntiles, int* __restrict__ tile_e,
                                 int* __restrict__ tile_r, int* __restrict__ token_list) {
  if (threadIdx.x != 0 || blockIdx.x != 0) return;
  int off = 0, t = 0;
  for (int e = 0; e < NE; ++e) {
    int c = counts[e];
    cursors[e] = off;
    int padded = (c + TILE_M - 1) / TILE_M * TILE_M;
    for (int i = 0; i < padded / TILE_M; ++i) {
      tile_e[t] = e;
      tile_r[t] = off + i * TILE_M;
      ++t;
    }
    for (int i = c; i < padded; ++i) token_list[off + i] = -1;  // sentinel pad
    off += padded;
  }
  *ntiles = t;
}

// ---------------- kernel 3: scatter tokens into expert groups ----------------
__global__ __launch_bounds__(256) void moe_scatter_kernel(
    const int* __restrict__ eidx, int* __restrict__ cursors,
    int* __restrict__ token_list, int* __restrict__ pos_of) {
  int t = blockIdx.x * 256 + threadIdx.x;
  if (t >= T_TOK) return;
#pragma unroll
  for (int s = 0; s < 2; ++s) {
    int e = eidx[2 * t + s];
    int p = atomicAdd(&cursors[e], 1);
    token_list[p] = t;
    pos_of[2 * t + s] = p;
  }
}

// ---------------- kernel 4: x fp32 -> packed bf16 ----------------
__global__ __launch_bounds__(256) void moe_cvt_x_kernel(const float* __restrict__ x,
                                                        unsigned int* __restrict__ xb) {
  int g = blockIdx.x * 256 + threadIdx.x;   // over T*H/2
  const float2 v = reinterpret_cast<const float2*>(x)[g];
  xb[g] = pack_bf16x2(v.x, v.y);
}

// ---------------- kernel 5: fused gate+up GEMM + SwiGLU epilogue ----------------
// grid (IN/TILE_N, MAX_TILES), block 512 (16 waves, 4M x 4N), wave tile 32x32, 2 acc
// sets. Software-pipelined: weights prefetched to VGPRs, x tile async double-buffered.
__global__ __launch_bounds__(512) void moe_gateup_kernel(
    const unsigned int* __restrict__ xb,  // [T][H/2] bf16 pairs
    const float* __restrict__ Wg, const float* __restrict__ Wu,
    const int* __restrict__ tile_e, const int* __restrict__ tile_r,
    const int* __restrict__ ntiles, const int* __restrict__ token_list,
    unsigned short* __restrict__ act) {
  int tile = blockIdx.y;
  if (tile >= *ntiles) return;
  int e = tile_e[tile];
  int row0 = tile_r[tile];
  int n0 = blockIdx.x * TILE_N;

  __shared__ __align__(16) unsigned int xt[2][TILE_M][LDW];
  __shared__ __align__(16) unsigned int wgt[TILE_N][LDW];
  __shared__ __align__(16) unsigned int wut[TILE_N][LDW];
  __shared__ int sh_tok[TILE_M];

  int tid = threadIdx.x;
  int wave = tid >> 5, lane = tid & 31;
  int r = lane & 15, hl = lane >> 4;
  int m0 = (wave & 3) * 32;   // wave M offset
  int nw = (wave >> 2) * 32;  // wave N offset

  if (tid < TILE_M) {
    int v = token_list[row0 + tid];
    if ((unsigned)v >= (unsigned)T_TOK) v = 0;  // clamp sentinel pads
    sh_tok[tid] = v;
  }
  __syncthreads();

  const float* wg_base = Wg + (size_t)e * IN * H;
  const float* wu_base = Wu + (size_t)e * IN * H;

  // staging roles: x tile 128 rows x 4 uint4 (1/thread); weights 128 rows x 8 float4 x2
  int xrow = tid >> 2, xq = tid & 3;
  size_t xgbase = (size_t)sh_tok[xrow] * (H / 2) + xq * 4;
  unsigned xlds[2] = {lds_off32(&xt[0][xrow][xq * 4]), lds_off32(&xt[1][xrow][xq * 4])};
  int wrow[2], wc4[2];
#pragma unroll
  for (int it = 0; it < 2; ++it) {
    int flat = it * 512 + tid;
    wrow[it] = flat >> 3;
    wc4[it] = flat & 7;
  }

  v8f gacc[2][2], uacc[2][2];
#pragma unroll
  for (int a = 0; a < 2; ++a)
#pragma unroll
    for (int b = 0; b < 2; ++b) { gacc[a][b] = zero8(); uacc[a][b] = zero8(); }

  // prologue: prefetch step 0
  float4 ga[2], ua[2];
  async_copy_b128(xlds[0], (const void*)(xb + xgbase));
#pragma unroll
  for (int it = 0; it < 2; ++it) {
    size_t gofs = (size_t)(n0 + wrow[it]) * H + wc4[it] * 4;
    ga[it] = *reinterpret_cast<const float4*>(wg_base + gofs);
    ua[it] = *reinterpret_cast<const float4*>(wu_base + gofs);
  }

  constexpr int NSTEP = H / BK;
  for (int s = 0; s < NSTEP; ++s) {
    int cur = s & 1;
    // dump prefetched weights to LDS (prev readers closed by trailing barrier)
#pragma unroll
    for (int it = 0; it < 2; ++it) {
      wgt[wrow[it]][wc4[it] * 2 + 0] = pack_bf16x2(ga[it].x, ga[it].y);
      wgt[wrow[it]][wc4[it] * 2 + 1] = pack_bf16x2(ga[it].z, ga[it].w);
      wut[wrow[it]][wc4[it] * 2 + 0] = pack_bf16x2(ua[it].x, ua[it].y);
      wut[wrow[it]][wc4[it] * 2 + 1] = pack_bf16x2(ua[it].z, ua[it].w);
    }
    wait_async0();
    __syncthreads();
    if (s + 1 < NSTEP) {  // prefetch step s+1 while WMMAs run
      int ksn = (s + 1) * BK;
      async_copy_b128(xlds[cur ^ 1], (const void*)(xb + xgbase + (ksn >> 1)));
#pragma unroll
      for (int it = 0; it < 2; ++it) {
        size_t gofs = (size_t)(n0 + wrow[it]) * H + ksn + wc4[it] * 4;
        ga[it] = *reinterpret_cast<const float4*>(wg_base + gofs);
        ua[it] = *reinterpret_cast<const float4*>(wu_base + gofs);
      }
    }

    // A fragments (16x32 bf16): two ds_load_b128 each
    FragBF af[2];
#pragma unroll
    for (int mf = 0; mf < 2; ++mf) {
      const uint4* p = reinterpret_cast<const uint4*>(&xt[cur][m0 + mf * 16 + r][0]);
      af[mf].q[0] = p[hl];
      af[mf].q[1] = p[2 + hl];
    }
#pragma unroll
    for (int nf = 0; nf < 2; ++nf) {
      int nl = nw + nf * 16 + r;
      FragBF bg, bu;
      const uint4* pg = reinterpret_cast<const uint4*>(&wgt[nl][0]);
      const uint4* pu = reinterpret_cast<const uint4*>(&wut[nl][0]);
      bg.q[0] = pg[hl]; bg.q[1] = pg[2 + hl];
      bu.q[0] = pu[hl]; bu.q[1] = pu[2 + hl];
#pragma unroll
      for (int mf = 0; mf < 2; ++mf) {
        gacc[mf][nf] = __builtin_amdgcn_wmma_f32_16x16x32_bf16(
            false, af[mf].v, false, bg.v, (short)0, gacc[mf][nf], false, false);
        uacc[mf][nf] = __builtin_amdgcn_wmma_f32_16x16x32_bf16(
            false, af[mf].v, false, bu.v, (short)0, uacc[mf][nf], false, false);
      }
    }
    __syncthreads();
  }

  // epilogue: act = silu(g) * u -> bf16, grouped-row layout
#pragma unroll
  for (int mf = 0; mf < 2; ++mf) {
#pragma unroll
    for (int nf = 0; nf < 2; ++nf) {
      int nc = n0 + nw + nf * 16 + r;
#pragma unroll
      for (int v = 0; v < 8; ++v) {
        int mrow = m0 + mf * 16 + v + 8 * hl;
        float a = silu_mul(gacc[mf][nf][v], uacc[mf][nf][v]);
        act[(size_t)(row0 + mrow) * IN + nc] = (unsigned short)f32_to_bf16_bits(a);
      }
    }
  }
}

// ---------------- kernel 6: down-projection GEMM ----------------
// grid (H/TILE_ND, MAX_TILES), block 512 (16 waves, 4M x 4N), wave tile 32x64.
// Software-pipelined like gate/up.
__global__ __launch_bounds__(512) void moe_down_kernel(
    const unsigned int* __restrict__ act,  // [PAD_CAP][IN/2] bf16 pairs, grouped
    const float* __restrict__ Wd,
    const int* __restrict__ tile_e, const int* __restrict__ tile_r,
    const int* __restrict__ ntiles, float* __restrict__ hout) {
  int tile = blockIdx.y;
  if (tile >= *ntiles) return;
  int e = tile_e[tile];
  int row0 = tile_r[tile];
  int n0 = blockIdx.x * TILE_ND;

  __shared__ __align__(16) unsigned int at[2][TILE_M][LDW];
  __shared__ __align__(16) unsigned int wdt[TILE_ND][LDW];

  int tid = threadIdx.x;
  int wave = tid >> 5, lane = tid & 31;
  int r = lane & 15, hl = lane >> 4;
  int m0 = (wave & 3) * 32;    // wave M offset
  int nw = (wave >> 2) * 64;   // wave N offset

  const float* wd_base = Wd + (size_t)e * H * IN;

  int xrow = tid >> 2, xq = tid & 3;  // act tile: 128 rows x 4 uint4
  size_t agbase = (size_t)(row0 + xrow) * (IN / 2) + xq * 4;
  unsigned alds[2] = {lds_off32(&at[0][xrow][xq * 4]), lds_off32(&at[1][xrow][xq * 4])};
  int wrow[4], wc4[4];
#pragma unroll
  for (int it = 0; it < 4; ++it) {
    int flat = it * 512 + tid;
    wrow[it] = flat >> 3;
    wc4[it] = flat & 7;
  }

  v8f acc[2][4];
#pragma unroll
  for (int a = 0; a < 2; ++a)
#pragma unroll
    for (int b = 0; b < 4; ++b) acc[a][b] = zero8();

  // prologue: prefetch step 0
  float4 da[4];
  async_copy_b128(alds[0], (const void*)(act + agbase));
#pragma unroll
  for (int it = 0; it < 4; ++it) {
    size_t gofs = (size_t)(n0 + wrow[it]) * IN + wc4[it] * 4;
    da[it] = *reinterpret_cast<const float4*>(wd_base + gofs);
  }

  constexpr int NSTEP = IN / BK;
  for (int s = 0; s < NSTEP; ++s) {
    int cur = s & 1;
#pragma unroll
    for (int it = 0; it < 4; ++it) {
      wdt[wrow[it]][wc4[it] * 2 + 0] = pack_bf16x2(da[it].x, da[it].y);
      wdt[wrow[it]][wc4[it] * 2 + 1] = pack_bf16x2(da[it].z, da[it].w);
    }
    wait_async0();
    __syncthreads();
    if (s + 1 < NSTEP) {
      int ksn = (s + 1) * BK;
      async_copy_b128(alds[cur ^ 1], (const void*)(act + agbase + (ksn >> 1)));
#pragma unroll
      for (int it = 0; it < 4; ++it) {
        size_t gofs = (size_t)(n0 + wrow[it]) * IN + ksn + wc4[it] * 4;
        da[it] = *reinterpret_cast<const float4*>(wd_base + gofs);
      }
    }

    FragBF af[2];
#pragma unroll
    for (int mf = 0; mf < 2; ++mf) {
      const uint4* p = reinterpret_cast<const uint4*>(&at[cur][m0 + mf * 16 + r][0]);
      af[mf].q[0] = p[hl];
      af[mf].q[1] = p[2 + hl];
    }
#pragma unroll
    for (int nf = 0; nf < 4; ++nf) {
      int nl = nw + nf * 16 + r;
      FragBF bw;
      const uint4* pw = reinterpret_cast<const uint4*>(&wdt[nl][0]);
      bw.q[0] = pw[hl]; bw.q[1] = pw[2 + hl];
#pragma unroll
      for (int mf = 0; mf < 2; ++mf) {
        acc[mf][nf] = __builtin_amdgcn_wmma_f32_16x16x32_bf16(
            false, af[mf].v, false, bw.v, (short)0, acc[mf][nf], false, false);
      }
    }
    __syncthreads();
  }

#pragma unroll
  for (int mf = 0; mf < 2; ++mf) {
#pragma unroll
    for (int nf = 0; nf < 4; ++nf) {
      int nc = n0 + nw + nf * 16 + r;
#pragma unroll
      for (int v = 0; v < 8; ++v) {
        int mrow = m0 + mf * 16 + v + 8 * hl;
        hout[(size_t)(row0 + mrow) * H + nc] = acc[mf][nf][v];
      }
    }
  }
}

// ---------------- kernel 7: deterministic per-token combine ----------------
__global__ __launch_bounds__(256) void moe_combine_kernel(
    const float* __restrict__ hout, const float* __restrict__ cw,
    const int* __restrict__ pos_of, float* __restrict__ out) {
  int g = blockIdx.x * 256 + threadIdx.x;  // over T*H
  int t = g >> 11;                          // H == 2048
  int h = g & (H - 1);
  float w0 = cw[2 * t + 0], w1 = cw[2 * t + 1];
  int p0 = pos_of[2 * t + 0], p1 = pos_of[2 * t + 1];
  out[g] = w0 * hout[(size_t)p0 * H + h] + w1 * hout[(size_t)p1 * H + h];
}

// ---------------- host launcher ----------------
extern "C" void kernel_launch(void* const* d_in, const int* in_sizes, int n_in,
                              void* d_out, int out_size, void* d_ws, size_t ws_size,
                              hipStream_t stream) {
  (void)in_sizes; (void)n_in; (void)out_size; (void)ws_size;
  const float* x  = (const float*)d_in[0];
  const float* rw = (const float*)d_in[1];
  const float* rb = (const float*)d_in[2];
  const float* Wg = (const float*)d_in[3];
  const float* Wu = (const float*)d_in[4];
  const float* Wd = (const float*)d_in[5];
  float* out = (float*)d_out;

  char* ws = (char*)d_ws;
  float* cw            = (float*)(ws + OFF_CW);
  int*   eidx          = (int*)(ws + OFF_EIDX);
  int*   pos_of        = (int*)(ws + OFF_POS);
  int*   counts        = (int*)(ws + OFF_CNT);
  int*   cursors       = (int*)(ws + OFF_CUR);
  int*   ntiles        = (int*)(ws + OFF_NT);
  int*   tile_e        = (int*)(ws + OFF_TE);
  int*   tile_r        = (int*)(ws + OFF_TR);
  int*   token_list    = (int*)(ws + OFF_TL);
  unsigned int*  xb    = (unsigned int*)(ws + OFF_XB);
  unsigned short* act  = (unsigned short*)(ws + OFF_ACT);
  float* hout          = (float*)(ws + OFF_HOUT);

  moe_init_kernel<<<1, 64, 0, stream>>>(counts);
  moe_router_kernel<<<T_TOK / 8, 256, 0, stream>>>(x, rw, rb, cw, eidx, counts);
  moe_setup_kernel<<<1, 1, 0, stream>>>(counts, cursors, ntiles, tile_e, tile_r, token_list);
  moe_scatter_kernel<<<T_TOK / 256, 256, 0, stream>>>(eidx, cursors, token_list, pos_of);
  moe_cvt_x_kernel<<<(T_TOK * H / 2) / 256, 256, 0, stream>>>(x, xb);
  moe_gateup_kernel<<<dim3(IN / TILE_N, MAX_TILES), 512, 0, stream>>>(
      xb, Wg, Wu, tile_e, tile_r, ntiles, token_list, act);
  moe_down_kernel<<<dim3(H / TILE_ND, MAX_TILES), 512, 0, stream>>>(
      (const unsigned int*)act, Wd, tile_e, tile_r, ntiles, hout);
  moe_combine_kernel<<<(T_TOK * H) / 256, 256, 0, stream>>>(hout, cw, pos_of, out);
}
